// HyperGAT_24180665877101
// MI455X (gfx1250) — matrix operand order repaired and checked by reading, not compiled, not run on hardware
//
#include <hip/hip_runtime.h>
#include <hip/hip_bf16.h>

typedef __attribute__((ext_vector_type(16))) __bf16 v16bf;
typedef __attribute__((ext_vector_type(8)))  float  v8f;

#define NEG_SLOPE 0.2f

// ---------- device helpers ----------
__device__ __forceinline__ unsigned encf(float f) {
    unsigned u = __float_as_uint(f);
    return (u & 0x80000000u) ? ~u : (u | 0x80000000u);
}
__device__ __forceinline__ float decf(unsigned u) {
    return (u & 0x80000000u) ? __uint_as_float(u & 0x7fffffffu)
                             : __uint_as_float(~u);
}
__device__ __forceinline__ void atomAddF(float* p, float v) {
    __hip_atomic_fetch_add(p, v, __ATOMIC_RELAXED, __HIP_MEMORY_SCOPE_AGENT);
}

// ---------- fp32 -> bf16 conversion (optionally with ELU) ----------
__global__ __launch_bounds__(256) void cvt_bf16_kernel(const float* __restrict__ in,
                                                       __bf16* __restrict__ out,
                                                       long long count, int do_elu) {
    long long tid = (long long)blockIdx.x * 256 + threadIdx.x;
    if (tid >= count) return;
    float x = in[tid];
    if (do_elu) x = (x > 0.0f) ? x : (__expf(x) - 1.0f);
    out[tid] = (__bf16)x;
}

// ---------- WMMA GEMM: hp[n][h*64+o] = sum_k hbf[n][k] * w[h][k][o] ----------
// Block: 256 threads = 8 waves; each wave owns a 16-row strip (block covers 128 rows).
// Weights for one head (256x64) are staged transposed in LDS as bf16: WT[o][k].
__global__ __launch_bounds__(256) void gemm_hp_wmma(const __bf16* __restrict__ hbf,
                                                    const float* __restrict__ w, // [4,256,64]
                                                    float* __restrict__ hp,      // [N,256]
                                                    int N) {
    __shared__ __bf16 WT[64 * 256]; // 32 KB: [col_local][k]

    const int t    = threadIdx.x;
    const int lane = t & 31;
    const int wave = t >> 5;
    const int row0 = blockIdx.x * 128 + wave * 16;
    const int m    = lane & 15;
    const bool hi  = lane >= 16;

    // --- preload A: 16 rows x 256 K, per-lane layout per ISA 16-bit A 16x32 table ---
    int arow = row0 + m;
    if (arow >= N) arow = N - 1; // clamp: garbage rows never stored
    const __bf16* arowp = hbf + (size_t)arow * 256;
    union AB { v16bf v; uint4 q[2]; };
    AB a[8];
    const int abase = hi ? 8 : 0;
#pragma unroll
    for (int kc = 0; kc < 8; ++kc) {
        const int k1 = kc * 32 + abase;
        a[kc].q[0] = *(const uint4*)(arowp + k1);        // K = base..base+7
        a[kc].q[1] = *(const uint4*)(arowp + k1 + 16);   // K = base+16..base+23
    }

    const bool full_tile = (row0 + 15) < N;

#pragma unroll 1
    for (int g = 0; g < 4; ++g) { // head
        __syncthreads();
        // cooperative transpose-load of head g weights into LDS (fp32 -> bf16)
        const float* wg = w + (size_t)g * 256 * 64;
#pragma unroll 1
        for (int i = 0; i < 64; ++i) {
            int k = i * 4 + (t >> 6);
            int o = t & 63;
            WT[o * 256 + k] = (__bf16)wg[k * 64 + o];
        }
        __syncthreads();

        // prefetch next head's fp32 weight panel while this head computes
        if (g < 3) {
            const float* wn = w + (size_t)(g + 1) * 256 * 64;
            __builtin_prefetch(wn + (size_t)t * 64, 0, 1);
        }

#pragma unroll 1
        for (int ct = 0; ct < 4; ++ct) { // 16-col tiles within head
            const int col_local = ct * 16 + m;
            const __bf16* bcol = &WT[col_local * 256 + (hi ? 8 : 0)];
            // batch-load all 8 B chunks so the WMMAs can issue back-to-back
            AB b[8];
#pragma unroll
            for (int kc = 0; kc < 8; ++kc) {
                b[kc].q[0] = *(const uint4*)(bcol + kc * 32);     // K = koff..koff+7
                b[kc].q[1] = *(const uint4*)(bcol + kc * 32 + 16); // K = koff+16..koff+23
            }
            v8f acc = {};
#pragma unroll
            for (int kc = 0; kc < 8; ++kc) {
                acc = __builtin_amdgcn_wmma_f32_16x16x32_bf16(
                    false, a[kc].v, false, b[kc].v, (short)0, acc, false, false);
            }
            const int col = g * 64 + ct * 16 + m;
            float* colp = hp + col;
            const int rbase = row0 + (hi ? 8 : 0);
            if (full_tile) {
#pragma unroll
                for (int j = 0; j < 8; ++j)
                    colp[(size_t)(rbase + j) * 256] = acc[j];
            } else {
#pragma unroll
                for (int j = 0; j < 8; ++j) {
                    int r = rbase + j;
                    if (r < N) colp[(size_t)r * 256] = acc[j];
                }
            }
        }
    }
}

// ---------- attention logits: asb/atb[n][h] = <hp[n][h*64:...], a_src/trg[h]> ----------
__global__ __launch_bounds__(256) void attn_kernel(const float* __restrict__ hp,
                                                   const float* __restrict__ a_src,
                                                   const float* __restrict__ a_trg,
                                                   float* __restrict__ asb,
                                                   float* __restrict__ atb, int N) {
    int tid = blockIdx.x * 256 + threadIdx.x;
    if (tid >= N * 4) return;
    int n = tid >> 2, h = tid & 3;
    const float* row = hp + (size_t)n * 256 + h * 64;
    const float* vs = a_src + h * 64;
    const float* vt = a_trg + h * 64;
    float s = 0.0f, tt = 0.0f;
#pragma unroll
    for (int k = 0; k < 64; k += 4) {
        float4 r = *(const float4*)(row + k);
        float4 x = *(const float4*)(vs + k);
        float4 y = *(const float4*)(vt + k);
        s  += r.x * x.x + r.y * x.y + r.z * x.z + r.w * x.w;
        tt += r.x * y.x + r.y * y.y + r.z * y.z + r.w * y.w;
    }
    asb[tid] = s;
    atb[tid] = tt;
}

// ---------- edge pass 1: e = leaky_relu(as[src]+at[trg]); global max ----------
__global__ __launch_bounds__(256) void edge_e_max(const int* __restrict__ src,
                                                  const int* __restrict__ trg,
                                                  const float* __restrict__ asb,
                                                  const float* __restrict__ atb,
                                                  float* __restrict__ ebuf,
                                                  unsigned* __restrict__ maxenc, int E) {
    __shared__ float sm[256];
    int tid = blockIdx.x * 256 + threadIdx.x;
    float e = -3.4e38f;
    if (tid < E * 4) {
        int edge = tid >> 2, h = tid & 3;
        float v = asb[src[edge] * 4 + h] + atb[trg[edge] * 4 + h];
        e = (v > 0.0f) ? v : NEG_SLOPE * v;
        ebuf[tid] = e;
    }
    sm[threadIdx.x] = e;
    __syncthreads();
#pragma unroll
    for (int s = 128; s > 0; s >>= 1) {
        if (threadIdx.x < s) sm[threadIdx.x] = fmaxf(sm[threadIdx.x], sm[threadIdx.x + s]);
        __syncthreads();
    }
    if (threadIdx.x == 0) atomicMax(maxenc, encf(sm[0]));
}

// ---------- edge pass 2: exp(e-max), segment-sum denominators ----------
__global__ __launch_bounds__(256) void edge_exp_denom(const int* __restrict__ trg,
                                                      float* __restrict__ ebuf,
                                                      const unsigned* __restrict__ maxenc,
                                                      float* __restrict__ denom, int E) {
    int tid = blockIdx.x * 256 + threadIdx.x;
    if (tid >= E * 4) return;
    float mx = decf(*maxenc);
    int edge = tid >> 2, h = tid & 3;
    float ex = __expf(ebuf[tid] - mx);
    ebuf[tid] = ex;
    atomAddF(&denom[(size_t)trg[edge] * 4 + h], ex);
}

// ---------- edge pass 3: agg[trg] += hp[src] * attn ----------
__global__ __launch_bounds__(256) void edge_agg(const int* __restrict__ src,
                                                const int* __restrict__ trg,
                                                const float* __restrict__ hp,
                                                const float* __restrict__ ebuf,
                                                const float* __restrict__ denom,
                                                float* __restrict__ agg, int E) {
    long long tid = (long long)blockIdx.x * 256 + threadIdx.x;
    if (tid >= (long long)E * 64) return;
    int q = (int)(tid & 63);
    long long edge = tid >> 6;
    int h = q >> 4, c4 = (q & 15) * 4;
    int s = src[edge], d = trg[edge];
    float attn = ebuf[edge * 4 + h] / (denom[(size_t)d * 4 + h] + 1e-16f);
    float4 hv = *(const float4*)(hp + (size_t)s * 256 + h * 64 + c4);
    float* out = agg + (size_t)d * 256 + h * 64 + c4;
    atomAddF(out + 0, hv.x * attn);
    atomAddF(out + 1, hv.y * attn);
    atomAddF(out + 2, hv.z * attn);
    atomAddF(out + 3, hv.w * attn);
}

// ---------- mean over heads: out[n][o] = mean_h agg[n][h*64+o] ----------
__global__ __launch_bounds__(256) void mean_heads(const float* __restrict__ agg,
                                                  float* __restrict__ out, int N) {
    int tid = blockIdx.x * 256 + threadIdx.x;
    if (tid >= N * 64) return;
    int n = tid >> 6, o = tid & 63;
    const float* r = agg + (size_t)n * 256;
    out[tid] = 0.25f * (r[o] + r[64 + o] + r[128 + o] + r[192 + o]);
}

// ---------- final FC (128->16) + log-softmax ----------
__global__ __launch_bounds__(256) void fc_logsoftmax(const float* __restrict__ out_u,
                                                     const float* __restrict__ out_t,
                                                     const float* __restrict__ fcw, // [128,16]
                                                     const float* __restrict__ fcb,
                                                     float* __restrict__ out, int N) {
    int n = blockIdx.x * 256 + threadIdx.x;
    if (n >= N) return;
    float l[16];
#pragma unroll
    for (int j = 0; j < 16; ++j) l[j] = fcb[j];
    const float* ru = out_u + (size_t)n * 64;
    const float* rt = out_t + (size_t)n * 64;
    for (int k = 0; k < 64; ++k) {
        float v = ru[k];
        const float* wr = fcw + k * 16;
#pragma unroll
        for (int j = 0; j < 16; ++j) l[j] += v * wr[j];
    }
    for (int k = 0; k < 64; ++k) {
        float v = rt[k];
        const float* wr = fcw + (64 + k) * 16;
#pragma unroll
        for (int j = 0; j < 16; ++j) l[j] += v * wr[j];
    }
    float m = l[0];
#pragma unroll
    for (int j = 1; j < 16; ++j) m = fmaxf(m, l[j]);
    float sum = 0.0f;
#pragma unroll
    for (int j = 0; j < 16; ++j) sum += __expf(l[j] - m);
    float lse = __logf(sum);
    float* orow = out + (size_t)n * 16;
#pragma unroll
    for (int j = 0; j < 16; ++j) orow[j] = l[j] - m - lse;
}

// ==================== host orchestration ====================
namespace {

struct Ws {
    __bf16* hbf;     // [N,256] bf16 activations
    float*  hp;      // [N,256]
    float*  agg;     // [N,256]
    float*  asb;     // [N,4]
    float*  atb;     // [N,4]
    float*  denom;   // [N,4]
    float*  ebuf;    // [E,4]
    float*  out_u;   // [N,64]
    float*  out_t;   // [N,64]
    unsigned* maxenc;
};

inline unsigned cdiv(long long a, long long b) { return (unsigned)((a + b - 1) / b); }

void run_stack(const float* emb, const int* src, const int* trg,
               const float* w0, const float* as0, const float* at0,
               const float* w1, const float* as1, const float* at1,
               float* out_side, const Ws& W, int N, int E, hipStream_t stream) {
    long long nfeat = (long long)N * 256;
    cvt_bf16_kernel<<<cdiv(nfeat, 256), 256, 0, stream>>>(emb, W.hbf, nfeat, 0);

    for (int layer = 0; layer < 2; ++layer) {
        const float* w  = layer ? w1  : w0;
        const float* as = layer ? as1 : as0;
        const float* at = layer ? at1 : at0;

        gemm_hp_wmma<<<cdiv(N, 128), 256, 0, stream>>>(W.hbf, w, W.hp, N);
        attn_kernel<<<cdiv((long long)N * 4, 256), 256, 0, stream>>>(W.hp, as, at, W.asb, W.atb, N);

        hipMemsetAsync(W.maxenc, 0, sizeof(unsigned), stream);
        hipMemsetAsync(W.denom, 0, (size_t)N * 4 * sizeof(float), stream);
        hipMemsetAsync(W.agg, 0, (size_t)N * 256 * sizeof(float), stream);

        edge_e_max<<<cdiv((long long)E * 4, 256), 256, 0, stream>>>(src, trg, W.asb, W.atb, W.ebuf, W.maxenc, E);
        edge_exp_denom<<<cdiv((long long)E * 4, 256), 256, 0, stream>>>(trg, W.ebuf, W.maxenc, W.denom, E);
        edge_agg<<<cdiv((long long)E * 64, 256), 256, 0, stream>>>(src, trg, W.hp, W.ebuf, W.denom, W.agg, E);

        if (layer == 0)
            cvt_bf16_kernel<<<cdiv(nfeat, 256), 256, 0, stream>>>(W.agg, W.hbf, nfeat, 1);
    }
    mean_heads<<<cdiv((long long)N * 64, 256), 256, 0, stream>>>(W.agg, out_side, N);
}

} // namespace

extern "C" void kernel_launch(void* const* d_in, const int* in_sizes, int n_in,
                              void* d_out, int out_size, void* d_ws, size_t ws_size,
                              hipStream_t stream) {
    const int N = in_sizes[0] / 256;  // 20000
    const int E = in_sizes[2];        // 320000

    const float* emb_u = (const float*)d_in[0];
    const float* emb_t = (const float*)d_in[1];
    const int* src_u = (const int*)d_in[2];
    const int* trg_u = (const int*)d_in[3];
    const int* src_t = (const int*)d_in[4];
    const int* trg_t = (const int*)d_in[5];
    const float* w_u0 = (const float*)d_in[6];
    const float* a_src_u0 = (const float*)d_in[7];
    const float* a_trg_u0 = (const float*)d_in[8];
    const float* w_u1 = (const float*)d_in[9];
    const float* a_src_u1 = (const float*)d_in[10];
    const float* a_trg_u1 = (const float*)d_in[11];
    const float* w_t0 = (const float*)d_in[12];
    const float* a_src_t0 = (const float*)d_in[13];
    const float* a_trg_t0 = (const float*)d_in[14];
    const float* w_t1 = (const float*)d_in[15];
    const float* a_src_t1 = (const float*)d_in[16];
    const float* a_trg_t1 = (const float*)d_in[17];
    const float* fc_w = (const float*)d_in[18];
    const float* fc_b = (const float*)d_in[19];

    // carve workspace (256B-aligned chunks)
    char* p = (char*)d_ws;
    auto carve = [&](size_t bytes) -> void* {
        void* r = (void*)p;
        p += (bytes + 255) & ~(size_t)255;
        return r;
    };
    Ws W;
    W.hbf    = (__bf16*)  carve((size_t)N * 256 * sizeof(__bf16));
    W.hp     = (float*)   carve((size_t)N * 256 * sizeof(float));
    W.agg    = (float*)   carve((size_t)N * 256 * sizeof(float));
    W.asb    = (float*)   carve((size_t)N * 4 * sizeof(float));
    W.atb    = (float*)   carve((size_t)N * 4 * sizeof(float));
    W.denom  = (float*)   carve((size_t)N * 4 * sizeof(float));
    W.ebuf   = (float*)   carve((size_t)E * 4 * sizeof(float));
    W.out_u  = (float*)   carve((size_t)N * 64 * sizeof(float));
    W.out_t  = (float*)   carve((size_t)N * 64 * sizeof(float));
    W.maxenc = (unsigned*)carve(256);

    run_stack(emb_u, src_u, trg_u, w_u0, a_src_u0, a_trg_u0,
              w_u1, a_src_u1, a_trg_u1, W.out_u, W, N, E, stream);
    run_stack(emb_t, src_t, trg_t, w_t0, a_src_t0, a_trg_t0,
              w_t1, a_src_t1, a_trg_t1, W.out_t, W, N, E, stream);

    fc_logsoftmax<<<cdiv(N, 256), 256, 0, stream>>>(W.out_u, W.out_t, fc_w, fc_b,
                                                    (float*)d_out, N);
}
